// MoELayer_51848845197420
// MI455X (gfx1250) — matrix operand order repaired and checked by reading, not compiled, not run on hardware
//
#include <hip/hip_runtime.h>

// ---------------------------------------------------------------------------
// MoE top-2 routed FFN for MI455X (gfx1250, wave32, WMMA bf16).
//   x[N=8192,C=1024] f32, Wg[C,E=8], W1[E,C,F=4096], W2[E,F,C] -> out[N,C]
// Pipeline: gate/top2 -> prefix -> x/W bf16 conversion -> GEMM1(H=relu(XW1))
//           -> GEMM2(out += w * H W2), expert-gathered, M-tile=128.
// GEMM1 stages the gathered X tile with GLOBAL_LOAD_ASYNC_TO_LDS_B128
// (ASYNCcnt-tracked, drained with s_wait_asynccnt before the barrier).
// ---------------------------------------------------------------------------

typedef __attribute__((ext_vector_type(16))) __bf16 v16bf;
typedef __attribute__((ext_vector_type(8)))  __bf16 v8bf;
typedef __attribute__((ext_vector_type(4)))  __bf16 v4bf;
typedef __attribute__((ext_vector_type(8)))  float  v8f;

#define MOE_C  1024
#define MOE_F  4096
#define MOE_E  8
#define MOE_MT 128            // token M-tile for both GEMMs

static __device__ __forceinline__ __bf16 f2bf(float f) {
    unsigned u = __builtin_bit_cast(unsigned, f);
    u += 0x7fffu + ((u >> 16) & 1u);            // round-to-nearest-even
    unsigned short s = (unsigned short)(u >> 16);
    return __builtin_bit_cast(__bf16, s);
}

// ---------------------------------------------------------------------------
// Gating: one wave32 per token. logits = x @ Wg, softmax(8), top-2, renorm,
// atomic-append into per-expert gather lists.
// ---------------------------------------------------------------------------
__global__ __launch_bounds__(256) void moe_gate_topk(
    const float* __restrict__ x, const float* __restrict__ Wg,
    int* __restrict__ counts, int* __restrict__ tok_idx,
    float* __restrict__ tok_w, int N)
{
    const int lane = threadIdx.x & 31;
    const int wave = threadIdx.x >> 5;
    const int n = blockIdx.x * 8 + wave;
    if (n >= N) return;

    float acc[MOE_E];
#pragma unroll
    for (int e = 0; e < MOE_E; ++e) acc[e] = 0.0f;

    const float* xr = x + (size_t)n * MOE_C;
    for (int c = lane; c < MOE_C; c += 32) {
        float xv = xr[c];
        const float4* wg4 = (const float4*)(Wg + (size_t)c * MOE_E);
        float4 w0 = wg4[0], w1 = wg4[1];
        acc[0] += xv * w0.x; acc[1] += xv * w0.y;
        acc[2] += xv * w0.z; acc[3] += xv * w0.w;
        acc[4] += xv * w1.x; acc[5] += xv * w1.y;
        acc[6] += xv * w1.z; acc[7] += xv * w1.w;
    }
#pragma unroll
    for (int e = 0; e < MOE_E; ++e) {
#pragma unroll
        for (int off = 16; off > 0; off >>= 1)
            acc[e] += __shfl_xor(acc[e], off, 32);
    }

    if (lane == 0) {
        float mx = acc[0];
#pragma unroll
        for (int e = 1; e < MOE_E; ++e) mx = fmaxf(mx, acc[e]);
        float p[MOE_E], s = 0.0f;
#pragma unroll
        for (int e = 0; e < MOE_E; ++e) { p[e] = __expf(acc[e] - mx); s += p[e]; }
        int i0 = 0;
#pragma unroll
        for (int e = 1; e < MOE_E; ++e) if (p[e] > p[i0]) i0 = e;
        int i1 = (i0 == 0) ? 1 : 0;
#pragma unroll
        for (int e = 0; e < MOE_E; ++e)
            if (e != i0 && p[e] > p[i1]) i1 = e;
        float w0 = p[i0] / s, w1 = p[i1] / s;
        float t = w0 + w1 + 1e-9f;
        w0 /= t; w1 /= t;

        int p0 = atomicAdd(&counts[i0], 1);
        tok_idx[(size_t)i0 * N + p0] = n;  tok_w[(size_t)i0 * N + p0] = w0;
        int p1 = atomicAdd(&counts[i1], 1);
        tok_idx[(size_t)i1 * N + p1] = n;  tok_w[(size_t)i1 * N + p1] = w1;
    }
}

// Exclusive prefix over 8 expert counts (tiny).
__global__ void moe_prefix(const int* __restrict__ counts, int* __restrict__ base) {
    if (threadIdx.x == 0) {
        int s = 0;
#pragma unroll
        for (int e = 0; e < MOE_E; ++e) { base[e] = s; s += counts[e]; }
    }
}

// x f32 -> xb bf16, coalesced (feeds the async LDS gather in GEMM1).
__global__ __launch_bounds__(256) void moe_x_bf16(
    const float* __restrict__ x, __bf16* __restrict__ xb, long total4)
{
    long i = (long)blockIdx.x * 256 + threadIdx.x;
    if (i >= total4) return;
    float4 v = *(const float4*)(x + i * 4);
    v4bf h; h[0] = f2bf(v.x); h[1] = f2bf(v.y); h[2] = f2bf(v.z); h[3] = f2bf(v.w);
    *(v4bf*)(xb + i * 4) = h;
}

// ---------------------------------------------------------------------------
// Tiled transpose + f32->bf16: in[z][R][Cc] -> out[z][Cc][R].  Block 32x8.
// ---------------------------------------------------------------------------
__global__ __launch_bounds__(256) void moe_w_transpose_bf16(
    const float* __restrict__ in, __bf16* __restrict__ out, int R, int Cc)
{
    __shared__ float tile[32][33];
    const int z = blockIdx.z;
    const float* src = in + (size_t)z * R * Cc;
    __bf16* dst = out + (size_t)z * Cc * R;
    const int c0 = blockIdx.x * 32, r0 = blockIdx.y * 32;
    const int tx = threadIdx.x, ty = threadIdx.y;
#pragma unroll
    for (int i = 0; i < 32; i += 8)
        tile[ty + i][tx] = src[(size_t)(r0 + ty + i) * Cc + (c0 + tx)];
    __syncthreads();
#pragma unroll
    for (int i = 0; i < 32; i += 8)
        dst[(size_t)(c0 + ty + i) * R + (r0 + tx)] = f2bf(tile[tx][ty + i]);
}

// ---------------------------------------------------------------------------
// GEMM1: H[base[e]+p, :F] = relu(Xg[p, :C] @ W1[e]).
// Block = 8 waves (2M x 4N), tile M=128 x N=128, K-step 32.
// X tile gathered into LDS with async global->LDS b128 copies (no VGPR
// round-trip); A frags ds_load_b128 pairs; B frags global b128 pairs from
// bf16 W1t[E][F][C].
// ---------------------------------------------------------------------------
__global__ __launch_bounds__(256, 1) void moe_gemm1(
    const __bf16* __restrict__ xb, const __bf16* __restrict__ W1t,
    const int* __restrict__ counts, const int* __restrict__ base,
    const int* __restrict__ tok_idx, __bf16* __restrict__ H, int N)
{
    __shared__ __bf16 Xlds[MOE_MT * MOE_C];   // 256 KB
    __shared__ int    tidx_s[MOE_MT];

    const int mtiles = N / MOE_MT;
    const int e  = blockIdx.x / mtiles;
    const int m0 = (blockIdx.x % mtiles) * MOE_MT;
    const int cnt = counts[e];
    if (m0 >= cnt) return;
    const int hbase = base[e];
    const int f0 = blockIdx.y * 128;

    const int tid = threadIdx.x;
    if (tid < MOE_MT) {
        int p = m0 + tid;
        tidx_s[tid] = (p < cnt) ? tok_idx[(size_t)e * N + p] : -1;
    }
    __syncthreads();

    // Gather X rows (bf16) into LDS: 16B chunks via async global->LDS copies.
    for (int idx = tid; idx < MOE_MT * (MOE_C / 8); idx += 256) {
        int row = idx >> 7, c = (idx & 127) * 8;
        int t = tidx_s[row];
        if (t >= 0) {
            const __bf16* src = xb + (size_t)t * MOE_C + c;
            unsigned ldsoff = (unsigned)(size_t)&Xlds[row * MOE_C + c];
            asm volatile("global_load_async_to_lds_b128 %0, %1, off"
                         :: "v"(ldsoff), "v"(src) : "memory");
        } else {
            v8bf z = (v8bf){};
            *(v8bf*)&Xlds[row * MOE_C + c] = z;     // zero-fill tail rows
        }
    }
    asm volatile("s_wait_asynccnt 0x0" ::: "memory");
    __syncthreads();

    const int lane  = tid & 31, wave = tid >> 5;
    const int wm    = wave & 1, wn = wave >> 1;       // 2 x 4 wave grid
    const int mstrip = wm * 64;
    const int nstrip = f0 + wn * 32;
    const int ncol  = lane & 15;
    const int khalf = lane >> 4;
    const int mbase = khalf * 8;

    v8f acc[4][2];
#pragma unroll
    for (int mi = 0; mi < 4; ++mi)
#pragma unroll
        for (int ni = 0; ni < 2; ++ni) acc[mi][ni] = (v8f){};

    for (int k0 = 0; k0 < MOE_C; k0 += 32) {
        v16bf a[4], b[2];
#pragma unroll
        for (int mi = 0; mi < 4; ++mi) {
            const __bf16* ap = &Xlds[(mstrip + mi * 16 + (lane & 15)) * MOE_C
                                     + k0 + khalf * 8];
            v8bf lo = *(const v8bf*)ap, hi = *(const v8bf*)(ap + 16);
#pragma unroll
            for (int i = 0; i < 8; ++i) { a[mi][i] = lo[i]; a[mi][8 + i] = hi[i]; }
        }
#pragma unroll
        for (int ni = 0; ni < 2; ++ni) {
            const __bf16* bp = W1t + ((size_t)e * MOE_F + nstrip + ni * 16 + ncol) * MOE_C
                                   + k0 + khalf * 8;
            v8bf lo = *(const v8bf*)bp, hi = *(const v8bf*)(bp + 16);
#pragma unroll
            for (int i = 0; i < 8; ++i) { b[ni][i] = lo[i]; b[ni][8 + i] = hi[i]; }
            if (k0 + 32 < MOE_C) __builtin_prefetch(bp + 32, 0, 1);
        }
#pragma unroll
        for (int mi = 0; mi < 4; ++mi)
#pragma unroll
            for (int ni = 0; ni < 2; ++ni)
                acc[mi][ni] = __builtin_amdgcn_wmma_f32_16x16x32_bf16(
                    false, a[mi], false, b[ni], (short)0, acc[mi][ni], false, false);
    }

    // ReLU -> bf16 -> compact H rows (guard p<cnt: slot cnt.. belongs to e+1).
#pragma unroll
    for (int mi = 0; mi < 4; ++mi)
#pragma unroll
        for (int ni = 0; ni < 2; ++ni)
#pragma unroll
            for (int r = 0; r < 8; ++r) {
                int lm = mstrip + mi * 16 + mbase + r;
                int p  = m0 + lm;
                if (p < cnt)
                    H[(size_t)(hbase + p) * MOE_F + nstrip + ni * 16 + ncol] =
                        f2bf(fmaxf(acc[mi][ni][r], 0.0f));
            }
}

// ---------------------------------------------------------------------------
// GEMM2: out[token] += w * (H[row,:F] @ W2[e]).  Tile M=128 x N=128, K=4096.
// H rows have K contiguous -> A frags are direct global b128 pairs (no LDS).
// B from bf16 W2t[E][C][F].  Atomic weighted scatter.
// ---------------------------------------------------------------------------
__global__ __launch_bounds__(256, 1) void moe_gemm2(
    const __bf16* __restrict__ H, const __bf16* __restrict__ W2t,
    const int* __restrict__ counts, const int* __restrict__ base,
    const int* __restrict__ tok_idx, const float* __restrict__ tok_w,
    float* __restrict__ out, int N)
{
    __shared__ int   tidx_s[MOE_MT];
    __shared__ float w_s[MOE_MT];

    const int mtiles = N / MOE_MT;
    const int e  = blockIdx.x / mtiles;
    const int m0 = (blockIdx.x % mtiles) * MOE_MT;
    const int cnt = counts[e];
    if (m0 >= cnt) return;
    const int hbase = base[e];
    const int c0 = blockIdx.y * 128;

    const int tid = threadIdx.x;
    if (tid < MOE_MT) {
        int p = m0 + tid;
        bool valid = (p < cnt);
        tidx_s[tid] = valid ? tok_idx[(size_t)e * N + p] : 0;
        w_s[tid]    = valid ? tok_w[(size_t)e * N + p]   : 0.0f;
    }
    __syncthreads();

    const int lane  = tid & 31, wave = tid >> 5;
    const int wm    = wave & 1, wn = wave >> 1;
    const int mstrip = wm * 64;
    const int nstrip = c0 + wn * 32;
    const int ncol  = lane & 15;
    const int khalf = lane >> 4;
    const int mbase = khalf * 8;

    v8f acc[4][2];
#pragma unroll
    for (int mi = 0; mi < 4; ++mi)
#pragma unroll
        for (int ni = 0; ni < 2; ++ni) acc[mi][ni] = (v8f){};

    for (int k0 = 0; k0 < MOE_F; k0 += 32) {
        v16bf a[4], b[2];
#pragma unroll
        for (int mi = 0; mi < 4; ++mi) {
            size_t hrow = (size_t)(hbase + m0 + mstrip + mi * 16 + (lane & 15));
            const __bf16* ap = H + hrow * MOE_F + k0 + khalf * 8;
            v8bf lo = *(const v8bf*)ap, hi = *(const v8bf*)(ap + 16);
#pragma unroll
            for (int i = 0; i < 8; ++i) { a[mi][i] = lo[i]; a[mi][8 + i] = hi[i]; }
        }
#pragma unroll
        for (int ni = 0; ni < 2; ++ni) {
            const __bf16* bp = W2t + ((size_t)e * MOE_C + nstrip + ni * 16 + ncol) * MOE_F
                                   + k0 + khalf * 8;
            v8bf lo = *(const v8bf*)bp, hi = *(const v8bf*)(bp + 16);
#pragma unroll
            for (int i = 0; i < 8; ++i) { b[ni][i] = lo[i]; b[ni][8 + i] = hi[i]; }
            if (k0 + 32 < MOE_F) __builtin_prefetch(bp + 32, 0, 1);
        }
#pragma unroll
        for (int mi = 0; mi < 4; ++mi)
#pragma unroll
            for (int ni = 0; ni < 2; ++ni)
                acc[mi][ni] = __builtin_amdgcn_wmma_f32_16x16x32_bf16(
                    false, a[mi], false, b[ni], (short)0, acc[mi][ni], false, false);
    }

#pragma unroll
    for (int mi = 0; mi < 4; ++mi)
#pragma unroll
        for (int ni = 0; ni < 2; ++ni)
#pragma unroll
            for (int r = 0; r < 8; ++r) {
                int lm = mstrip + mi * 16 + mbase + r;
                int p  = m0 + lm;
                if (p < cnt)
                    atomicAdd(&out[(size_t)tidx_s[lm] * MOE_C + nstrip + ni * 16 + ncol],
                              acc[mi][ni][r] * w_s[lm]);
            }
}

// ---------------------------------------------------------------------------
extern "C" void kernel_launch(void* const* d_in, const int* in_sizes, int n_in,
                              void* d_out, int out_size, void* d_ws, size_t ws_size,
                              hipStream_t stream) {
    const float* x  = (const float*)d_in[0];
    const float* Wg = (const float*)d_in[1];
    const float* W1 = (const float*)d_in[2];
    const float* W2 = (const float*)d_in[3];
    float* out = (float*)d_out;

    const int N = in_sizes[0] / MOE_C;         // 8192 tokens

    // ws layout
    char* ws = (char*)d_ws;
    int*   counts  = (int*)ws;                                   // [E]
    int*   basep   = (int*)(ws + 64);                            // [E]
    size_t off = 256;
    int*   tok_idx = (int*)(ws + off);    off += (size_t)MOE_E * N * sizeof(int);
    float* tok_w   = (float*)(ws + off);  off += (size_t)MOE_E * N * sizeof(float);
    off = (off + 255) & ~(size_t)255;
    __bf16* xb  = (__bf16*)(ws + off);    off += (size_t)N * MOE_C * 2;
    __bf16* W1t = (__bf16*)(ws + off);    off += (size_t)MOE_E * MOE_F * MOE_C * 2;
    __bf16* W2t = (__bf16*)(ws + off);    off += (size_t)MOE_E * MOE_C * MOE_F * 2;
    __bf16* H   = (__bf16*)(ws + off);    // (2N + MOE_MT) rows x F bf16

    hipMemsetAsync(ws, 0, 256, stream);
    hipMemsetAsync(out, 0, (size_t)out_size * sizeof(float), stream);

    moe_gate_topk<<<(N + 7) / 8, 256, 0, stream>>>(x, Wg, counts, tok_idx, tok_w, N);
    moe_prefix<<<1, 32, 0, stream>>>(counts, basep);

    const long total4 = (long)N * MOE_C / 4;
    moe_x_bf16<<<(unsigned)((total4 + 255) / 256), 256, 0, stream>>>(x, xb, total4);

    // W1[E][C][F] -> W1t[E][F][C];  W2[E][F][C] -> W2t[E][C][F]
    moe_w_transpose_bf16<<<dim3(MOE_F / 32, MOE_C / 32, MOE_E), dim3(32, 8), 0, stream>>>(
        W1, W1t, MOE_C, MOE_F);
    moe_w_transpose_bf16<<<dim3(MOE_C / 32, MOE_F / 32, MOE_E), dim3(32, 8), 0, stream>>>(
        W2, W2t, MOE_F, MOE_C);

    const int mtiles = N / MOE_MT;
    moe_gemm1<<<dim3(MOE_E * mtiles, MOE_F / 128), 256, 0, stream>>>(
        xb, W1t, counts, basep, tok_idx, H, N);
    moe_gemm2<<<dim3(MOE_E * mtiles, MOE_C / 128), 256, 0, stream>>>(
        H, W2t, counts, basep, tok_idx, tok_w, out, N);
}
